// AttnBlock_21612275433595
// MI455X (gfx1250) — matrix operand order repaired and checked by reading, not compiled, not run on hardware
//
#include <hip/hip_runtime.h>
#include <math.h>

#define DIM 512
#define L_SEQ 4096
#define B_SZ 8
#define ROWS (B_SZ * L_SEQ)      // 32768
#define NDEPTH 6
#define QKV_STRIDE (3 * DIM)     // 1536

typedef __attribute__((ext_vector_type(16))) __bf16 v16bf;
typedef __attribute__((ext_vector_type(8)))  float  v8f;
typedef __attribute__((ext_vector_type(4)))  unsigned int u32x4;

union Frag { v16bf v; u32x4 q[2]; };

__device__ inline v8f wmma_bf16(v16bf a, v16bf b, v8f c) {
    return __builtin_amdgcn_wmma_f32_16x16x32_bf16(false, a, false, b, (short)0, c, false, false);
}
__device__ inline v8f vzero8() { v8f z = {0.f,0.f,0.f,0.f,0.f,0.f,0.f,0.f}; return z; }

// Async 16B copy global -> LDS (per-lane), tracked with ASYNCcnt.
__device__ inline void async_copy16(void* lds, const void* gptr) {
    unsigned loff = (unsigned)(unsigned long long)(size_t)lds;   // flat->LDS low bits
    asm volatile("global_load_async_to_lds_b128 %0, %1, off"
                 :: "v"(loff), "v"(gptr)
                 : "memory");
}
__device__ inline void wait_async0() {
    asm volatile("s_wait_asynccnt 0" ::: "memory");
}

// ---------------------------------------------------------------------------
// fp32 -> bf16 weight conversion
// ---------------------------------------------------------------------------
__global__ void cvt_bf16_kernel(const float* __restrict__ x, __bf16* __restrict__ y, int n) {
    int i = blockIdx.x * 256 + threadIdx.x;
    if (i < n) y[i] = (__bf16)x[i];
}

// ---------------------------------------------------------------------------
// h[b][t][d] = x[b][d][t] + pos[t][d]   (one wave per token row)
// ---------------------------------------------------------------------------
__global__ __launch_bounds__(256)
void embed_kernel(const float* __restrict__ x, const float* __restrict__ pos,
                  float* __restrict__ h) {
    int wave = threadIdx.x >> 5, lane = threadIdx.x & 31;
    int row = blockIdx.x * 8 + wave;
    int b = row >> 12, t = row & 4095;
    for (int i = 0; i < 16; ++i) {
        int c = lane + i * 32;
        h[(size_t)row * DIM + c] =
            x[((size_t)(b * DIM + c)) * L_SEQ + t] + pos[(size_t)t * DIM + c];
    }
}

// ---------------------------------------------------------------------------
// LayerNorm over DIM=512, one wave per row. TRANSPOSE_OUT -> fp32 (B,DIM,L)
// else bf16 row-major (rows, DIM).
// ---------------------------------------------------------------------------
template <bool TRANSPOSE_OUT>
__global__ __launch_bounds__(256)
void ln_kernel(const float* __restrict__ x, const float* __restrict__ g,
               const float* __restrict__ bta, __bf16* __restrict__ yb,
               float* __restrict__ yf) {
    int wave = threadIdx.x >> 5, lane = threadIdx.x & 31;
    int row = blockIdx.x * 8 + wave;
    const float* p = x + (size_t)row * DIM;
    float vals[16]; float s = 0.f;
    for (int i = 0; i < 16; ++i) { vals[i] = p[lane + i * 32]; s += vals[i]; }
    for (int off = 1; off < 32; off <<= 1) s += __shfl_xor(s, off, 32);
    float mean = s * (1.f / DIM);
    float vs = 0.f;
    for (int i = 0; i < 16; ++i) { float d = vals[i] - mean; vs += d * d; }
    for (int off = 1; off < 32; off <<= 1) vs += __shfl_xor(vs, off, 32);
    float inv = rsqrtf(vs * (1.f / DIM) + 1e-5f);
    for (int i = 0; i < 16; ++i) {
        int c = lane + i * 32;
        float yv = (vals[i] - mean) * inv * g[c] + bta[c];
        if (TRANSPOSE_OUT) {
            int b = row >> 12, t = row & 4095;
            yf[((size_t)(b * DIM + c)) * L_SEQ + t] = yv;
        } else {
            yb[(size_t)row * DIM + c] = (__bf16)yv;
        }
    }
}

// ---------------------------------------------------------------------------
// Generic bf16 WMMA GEMM: out = epilogue(A[MxK] @ W[KxN] + bias (+res))
// 128x128x32 workgroup tile, 8 waves, each wave 64x32 (4x2 accum tiles).
// Double-buffered LDS; A tiles via GLOBAL_LOAD_ASYNC_TO_LDS_B128 (ASYNCcnt),
// B tiles software-pipelined through registers with transposed ds stores.
// ---------------------------------------------------------------------------
template <bool GELU, bool RES, bool OUTBF>
__global__ __launch_bounds__(256)
void gemm_wmma(const __bf16* __restrict__ A, const __bf16* __restrict__ W,
               const float* __restrict__ bias, const float* __restrict__ res,
               float* __restrict__ outF, __bf16* __restrict__ outB,
               int M, int N, int K) {
    __shared__ __align__(16) __bf16 As[2][128 * 32];   // [m][k]
    __shared__ __align__(16) __bf16 Bs[2][128 * 32];   // [n][k] (transposed stage)
    int tid  = threadIdx.x;
    int wave = tid >> 5, lane = tid & 31;
    int l = lane & 15, hl = lane >> 4;
    int wr = wave & 1, wc = wave >> 1;
    int rowBase = blockIdx.y * 128;
    int colBase = blockIdx.x * 128;

    v8f acc[4][2];
    for (int i = 0; i < 4; ++i) for (int j = 0; j < 2; ++j) acc[i][j] = vzero8();

    auto issueA = [&](int k0, int buf) {
        for (int it = 0; it < 2; ++it) {
            int vid = tid + it * 256;            // 0..511
            int r = vid >> 2, cv = vid & 3;
            async_copy16(&As[buf][r * 32 + cv * 8],
                         &A[(size_t)(rowBase + r) * K + k0 + cv * 8]);
        }
    };
    auto loadB = [&](int k0, u32x4* regs) {
        for (int it = 0; it < 2; ++it) {
            int vid = tid + it * 256;            // 0..511
            int kr = vid >> 4, nv = vid & 15;
            regs[it] = *(const u32x4*)&W[(size_t)(k0 + kr) * N + colBase + nv * 8];
        }
    };
    auto storeB = [&](const u32x4* regs, int buf) {
        for (int it = 0; it < 2; ++it) {
            int vid = tid + it * 256;
            int kr = vid >> 4, nv = vid & 15;
            __bf16 tmp[8]; *(u32x4*)tmp = regs[it];
            for (int e = 0; e < 8; ++e) Bs[buf][(nv * 8 + e) * 32 + kr] = tmp[e];
        }
    };

    int nk = K >> 5;
    {   // prologue: stage tile 0
        u32x4 b0[2];
        issueA(0, 0);
        loadB(0, b0);
        storeB(b0, 0);
        wait_async0();
        __syncthreads();
    }
    int cur = 0;
    for (int kt = 0; kt < nk; ++kt) {
        u32x4 bn[2];
        bool more = (kt + 1) < nk;
        if (more) {
            int k0n = (kt + 1) << 5;
            issueA(k0n, cur ^ 1);
            loadB(k0n, bn);
            if (kt + 2 < nk) {
                int k0p = (kt + 2) << 5;
                __builtin_prefetch(&W[(size_t)k0p * N + colBase], 0, 1);
                __builtin_prefetch(&A[(size_t)rowBase * K + k0p], 0, 1);
            }
        }

        // WMMA on current tile
        Frag bm[2];
        for (int j = 0; j < 2; ++j) {
            const u32x4* pb = (const u32x4*)&Bs[cur][(wc * 32 + j * 16 + l) * 32];
            bm[j].q[0] = pb[hl]; bm[j].q[1] = pb[2 + hl];
        }
        for (int i = 0; i < 4; ++i) {
            Frag a;
            const u32x4* pa = (const u32x4*)&As[cur][(wr * 64 + i * 16 + l) * 32];
            a.q[0] = pa[hl]; a.q[1] = pa[2 + hl];
            for (int j = 0; j < 2; ++j) acc[i][j] = wmma_bf16(a.v, bm[j].v, acc[i][j]);
        }

        if (more) {
            storeB(bn, cur ^ 1);
            wait_async0();
        }
        __syncthreads();
        cur ^= 1;
    }

    for (int i = 0; i < 4; ++i)
        for (int j = 0; j < 2; ++j) {
            int mrow = rowBase + wr * 64 + i * 16 + hl * 8;
            int ncol = colBase + wc * 32 + j * 16 + l;
            float bia = bias ? bias[ncol] : 0.f;
            for (int e = 0; e < 8; ++e) {
                float v = acc[i][j][e] + bia;
                if (GELU) v = 0.5f * v * (1.f + erff(v * 0.7071067811865476f));
                size_t idx = (size_t)(mrow + e) * N + ncol;
                if (RES) v += res[idx];
                if (OUTBF) outB[idx] = (__bf16)v;
                else       outF[idx] = v;
            }
        }
}

// ---------------------------------------------------------------------------
// Sinkhorn routing helpers
// ---------------------------------------------------------------------------
__global__ __launch_bounds__(64)
void bucket_mean_kernel(const __bf16* __restrict__ qkv, float* __restrict__ qm,
                        float* __restrict__ km) {
    int u = blockIdx.x, bh = blockIdx.y, d = threadIdx.x;
    int b = bh / 6, head = 2 + bh % 6;
    float sq = 0.f, sk = 0.f;
    size_t base = ((size_t)(b * L_SEQ + u * 64)) * QKV_STRIDE + head * 64 + d;
    for (int j = 0; j < 64; ++j) {
        sq += (float)qkv[base + (size_t)j * QKV_STRIDE];
        sk += (float)qkv[base + (size_t)j * QKV_STRIDE + DIM];
    }
    qm[(bh * 64 + u) * 64 + d] = sq * (1.f / 64.f);
    km[(bh * 64 + u) * 64 + d] = sk * (1.f / 64.f);
}

__global__ __launch_bounds__(64)
void routing_kernel(const float* __restrict__ qm, const float* __restrict__ km,
                    int* __restrict__ ridx, float* __restrict__ rwv) {
    int bh = blockIdx.x; int j = threadIdx.x;
    __shared__ float qml[64 * 64];
    __shared__ float kml[64 * 64];
    __shared__ float red[64];
    __shared__ int   redi[64];
    for (int i = j; i < 64 * 64; i += 64) {
        qml[i] = qm[(size_t)bh * 4096 + i];
        kml[i] = km[(size_t)bh * 4096 + i];
    }
    __syncthreads();
    for (int i = 0; i < 64; ++i) {
        float s = 0.f;
        for (int d = 0; d < 64; ++d) s += qml[i * 64 + d] * kml[j * 64 + d];
        s *= 0.125f;
        red[j] = s; redi[j] = j; __syncthreads();
        for (int st = 32; st > 0; st >>= 1) {
            if (j < st && red[j + st] > red[j]) { red[j] = red[j + st]; redi[j] = redi[j + st]; }
            __syncthreads();
        }
        float rmax = red[0]; int amax = redi[0]; __syncthreads();
        red[j] = __expf(s - rmax); __syncthreads();
        for (int st = 32; st > 0; st >>= 1) {
            if (j < st) red[j] += red[j + st];
            __syncthreads();
        }
        if (j == 0) { ridx[bh * 64 + i] = amax; rwv[bh * 64 + i] = 1.0f / red[0]; }
        __syncthreads();
    }
}

// ---------------------------------------------------------------------------
// Flash-style attention per window/bucket: 4 waves, WMMA for S and P@V.
// LOCAL: 192 keys (prev|cur|next windows), pad-masked.
// SINKHORN: 128 keys (weighted routed bucket | own bucket).
// Q (and valid LOCAL K rows) staged via async global->LDS copies.
// ---------------------------------------------------------------------------
template <int NKEYS, bool LOCAL>
__global__ __launch_bounds__(128)
void attn_kernel(const __bf16* __restrict__ qkv, const int* __restrict__ ridx,
                 const float* __restrict__ rwv, __bf16* __restrict__ out) {
    constexpr int SMEM_BYTES = 8192 + NKEYS * 256;   // Q + K + Vt
    __shared__ __align__(16) char smem[SMEM_BYTES];
    __bf16* Qs = (__bf16*)smem;                          // [64][64]
    __bf16* Ks = (__bf16*)(smem + 8192);                 // [NKEYS][64]
    __bf16* Vt = (__bf16*)(smem + 8192 + NKEYS * 128);   // [64][NKEYS]
    __bf16* Ps = (__bf16*)smem;                          // overlays Q|K after S

    int u = blockIdx.x;        // window / bucket
    int bh = blockIdx.y;
    int b, head;
    int t0 = u * 64;
    int rbucket = 0; float wgt = 1.f;
    if (LOCAL) { b = bh >> 1; head = bh & 1; }
    else {
        b = bh / 6; head = 2 + bh % 6;
        rbucket = ridx[bh * 64 + u];
        wgt = rwv[bh * 64 + u];
    }
    int tid = threadIdx.x;

    // ---- stage Q via async copies ----
    size_t qrow0 = ((size_t)(b * L_SEQ + t0)) * QKV_STRIDE + head * 64;
    for (int v = tid; v < 64 * 8; v += 128) {
        int r = v >> 3, cv = v & 7;
        async_copy16(&Qs[r * 64 + cv * 8],
                     &qkv[qrow0 + (size_t)r * QKV_STRIDE + cv * 8]);
    }
    // ---- stage K rows and V transposed ----
    for (int v = tid; v < NKEYS * 8; v += 128) {
        int j = v >> 3, cv = v & 7;
        int tK; bool valid = true; float sc = 1.f;
        if (LOCAL) { tK = t0 - 64 + j; valid = (tK >= 0) && (tK < L_SEQ); }
        else {
            if (j < 64) { tK = rbucket * 64 + j; sc = wgt; }
            else        { tK = t0 + (j - 64); }
        }
        size_t base = 0;
        if (valid) base = ((size_t)(b * L_SEQ + tK)) * QKV_STRIDE + head * 64 + cv * 8;

        // K rows
        if (LOCAL) {
            if (valid) {
                async_copy16(&Ks[j * 64 + cv * 8], &qkv[base + DIM]);
            } else {
                u32x4 z = {0u,0u,0u,0u};
                *(u32x4*)&Ks[j * 64 + cv * 8] = z;
            }
        } else {
            u32x4 kq = *(const u32x4*)&qkv[base + DIM];
            __bf16 kt[8]; *(u32x4*)kt = kq;
            if (j < 64)
                for (int e = 0; e < 8; ++e) kt[e] = (__bf16)((float)kt[e] * sc);
            *(u32x4*)&Ks[j * 64 + cv * 8] = *(u32x4*)kt;
        }

        // V transposed (always through registers)
        u32x4 vq = {0u,0u,0u,0u};
        if (valid) vq = *(const u32x4*)&qkv[base + 2 * DIM];
        __bf16 vt8[8]; *(u32x4*)vt8 = vq;
        if (!LOCAL && j < 64)
            for (int e = 0; e < 8; ++e) vt8[e] = (__bf16)((float)vt8[e] * sc);
        for (int e = 0; e < 8; ++e) Vt[(cv * 8 + e) * NKEYS + j] = vt8[e];
    }
    wait_async0();
    __syncthreads();

    constexpr int NT = NKEYS / 16;
    int lane = tid & 31, wv = tid >> 5;
    int l = lane & 15, hl = lane >> 4;
    int qr = wv * 16;

    // ---- S = Q @ K^T ----
    v8f accS[NT];
    for (int nt = 0; nt < NT; ++nt) accS[nt] = vzero8();
    for (int kk = 0; kk < 64; kk += 32) {
        Frag a;
        const u32x4* pa = (const u32x4*)&Qs[(qr + l) * 64 + kk];
        a.q[0] = pa[hl]; a.q[1] = pa[2 + hl];
        for (int nt = 0; nt < NT; ++nt) {
            Frag bm;
            const u32x4* pb = (const u32x4*)&Ks[(nt * 16 + l) * 64 + kk];
            bm.q[0] = pb[hl]; bm.q[1] = pb[2 + hl];
            accS[nt] = wmma_bf16(a.v, bm.v, accS[nt]);
        }
    }

    // ---- scale + mask + row softmax (16-lane shuffle reductions) ----
    float rsum[8];
    for (int nt = 0; nt < NT; ++nt)
        for (int e = 0; e < 8; ++e) {
            float s = accS[nt][e] * 0.125f;
            if (LOCAL) {
                int gk = t0 - 64 + nt * 16 + l;
                if (gk < 0 || gk >= L_SEQ) s = -1e30f;
            }
            accS[nt][e] = s;
        }
    for (int e = 0; e < 8; ++e) {
        float m = -1e30f;
        for (int nt = 0; nt < NT; ++nt) m = fmaxf(m, accS[nt][e]);
        for (int off = 1; off < 16; off <<= 1) m = fmaxf(m, __shfl_xor(m, off, 16));
        float sm = 0.f;
        for (int nt = 0; nt < NT; ++nt) {
            float p = __expf(accS[nt][e] - m);
            accS[nt][e] = p; sm += p;
        }
        for (int off = 1; off < 16; off <<= 1) sm += __shfl_xor(sm, off, 16);
        rsum[e] = sm;
    }
    __syncthreads();   // Q/K dead, safe to overlay
    for (int nt = 0; nt < NT; ++nt)
        for (int e = 0; e < 8; ++e)
            Ps[(qr + hl * 8 + e) * NKEYS + nt * 16 + l] = (__bf16)accS[nt][e];
    __syncthreads();

    // ---- O = P @ V ----
    v8f accO[4];
    for (int nt = 0; nt < 4; ++nt) accO[nt] = vzero8();
    for (int kk = 0; kk < NKEYS; kk += 32) {
        Frag a;
        const u32x4* pa = (const u32x4*)&Ps[(qr + l) * NKEYS + kk];
        a.q[0] = pa[hl]; a.q[1] = pa[2 + hl];
        for (int nt = 0; nt < 4; ++nt) {
            Frag bm;
            const u32x4* pb = (const u32x4*)&Vt[(nt * 16 + l) * NKEYS + kk];
            bm.q[0] = pb[hl]; bm.q[1] = pb[2 + hl];
            accO[nt] = wmma_bf16(a.v, bm.v, accO[nt]);
        }
    }
    for (int nt = 0; nt < 4; ++nt)
        for (int e = 0; e < 8; ++e) {
            int row = qr + hl * 8 + e;
            int t = t0 + row;
            int d = nt * 16 + l;
            float o = accO[nt][e] / rsum[e];
            out[((size_t)(b * L_SEQ + t)) * DIM + head * 64 + d] = (__bf16)o;
        }
}

// ---------------------------------------------------------------------------
extern "C" void kernel_launch(void* const* d_in, const int* in_sizes, int n_in,
                              void* d_out, int out_size, void* d_ws, size_t ws_size,
                              hipStream_t stream) {
    (void)in_sizes; (void)n_in; (void)out_size; (void)ws_size;
    const float* x    = (const float*)d_in[0];
    const float* pos  = (const float*)d_in[1];
    const float* Wqkv = (const float*)d_in[2];
    const float* Wout = (const float*)d_in[3];
    const float* bout = (const float*)d_in[4];
    const float* W1   = (const float*)d_in[5];
    const float* b1   = (const float*)d_in[6];
    const float* W2   = (const float*)d_in[7];
    const float* b2   = (const float*)d_in[8];
    const float* ln1g = (const float*)d_in[9];
    const float* ln1b = (const float*)d_in[10];
    const float* ln2g = (const float*)d_in[11];
    const float* ln2b = (const float*)d_in[12];
    const float* ng   = (const float*)d_in[13];
    const float* nbta = (const float*)d_in[14];

    char* ws = (char*)d_ws;
    size_t off = 0;
    auto alloc = [&](size_t bytes) {
        char* p = ws + off;
        off += (bytes + 255) & ~(size_t)255;
        return p;
    };
    float*  h    = (float*) alloc((size_t)ROWS * DIM * 4);
    __bf16* hn   = (__bf16*)alloc((size_t)ROWS * DIM * 2);
    __bf16* qkv  = (__bf16*)alloc((size_t)ROWS * 2048 * 2);   // shared with MLP mid
    __bf16* mid  = qkv;
    __bf16* aout = (__bf16*)alloc((size_t)ROWS * DIM * 2);
    __bf16* wqkv_bf = (__bf16*)alloc((size_t)NDEPTH * DIM * 1536 * 2);
    __bf16* wout_bf = (__bf16*)alloc((size_t)NDEPTH * DIM * DIM * 2);
    __bf16* w1_bf   = (__bf16*)alloc((size_t)NDEPTH * DIM * 2048 * 2);
    __bf16* w2_bf   = (__bf16*)alloc((size_t)NDEPTH * 2048 * DIM * 2);
    float*  qm   = (float*)alloc((size_t)48 * 64 * 64 * 4);
    float*  km   = (float*)alloc((size_t)48 * 64 * 64 * 4);
    int*    ridx = (int*)  alloc((size_t)48 * 64 * 4);
    float*  rwv  = (float*)alloc((size_t)48 * 64 * 4);

    auto cvt = [&](const float* s, __bf16* d, int n) {
        cvt_bf16_kernel<<<(n + 255) / 256, 256, 0, stream>>>(s, d, n);
    };
    cvt(Wqkv, wqkv_bf, NDEPTH * DIM * 1536);
    cvt(Wout, wout_bf, NDEPTH * DIM * DIM);
    cvt(W1,   w1_bf,   NDEPTH * DIM * 2048);
    cvt(W2,   w2_bf,   NDEPTH * 2048 * DIM);

    embed_kernel<<<ROWS / 8, 256, 0, stream>>>(x, pos, h);

    for (int i = 0; i < NDEPTH; ++i) {
        ln_kernel<false><<<ROWS / 8, 256, 0, stream>>>(h, ln1g + i * DIM, ln1b + i * DIM, hn, nullptr);
        gemm_wmma<false, false, true><<<dim3(1536 / 128, ROWS / 128), 256, 0, stream>>>(
            hn, wqkv_bf + (size_t)i * DIM * 1536, nullptr, nullptr, nullptr, qkv,
            ROWS, 1536, DIM);
        bucket_mean_kernel<<<dim3(64, 48), 64, 0, stream>>>(qkv, qm, km);
        routing_kernel<<<48, 64, 0, stream>>>(qm, km, ridx, rwv);
        attn_kernel<192, true ><<<dim3(64, 16), 128, 0, stream>>>(qkv, nullptr, nullptr, aout);
        attn_kernel<128, false><<<dim3(64, 48), 128, 0, stream>>>(qkv, ridx, rwv, aout);
        gemm_wmma<false, true, false><<<dim3(DIM / 128, ROWS / 128), 256, 0, stream>>>(
            aout, wout_bf + (size_t)i * DIM * DIM, bout + i * DIM, h, h, nullptr,
            ROWS, DIM, DIM);
        ln_kernel<false><<<ROWS / 8, 256, 0, stream>>>(h, ln2g + i * DIM, ln2b + i * DIM, hn, nullptr);
        gemm_wmma<true, false, true><<<dim3(2048 / 128, ROWS / 128), 256, 0, stream>>>(
            hn, w1_bf + (size_t)i * DIM * 2048, b1 + i * 2048, nullptr, nullptr, mid,
            ROWS, 2048, DIM);
        gemm_wmma<false, true, false><<<dim3(DIM / 128, ROWS / 128), 256, 0, stream>>>(
            mid, w2_bf + (size_t)i * 2048 * DIM, b2 + i * DIM, h, h, nullptr,
            ROWS, DIM, 2048);
    }
    ln_kernel<true><<<ROWS / 8, 256, 0, stream>>>(h, ng, nbta, nullptr, (float*)d_out);
}
